// SparseMoE_77506979824200
// MI455X (gfx1250) — matrix operand order repaired and checked by reading, not compiled, not run on hardware
//
#include <hip/hip_runtime.h>

// Problem sizes (match reference)
#define B_TOK 8192
#define DIM   1024
#define NEXP  8
#define HID   4096
#define TOPK  2

// GEMM tiling: 128x128 block tile, K-step 32, 256 threads = 8 wave32.
// Each wave computes a 32x64 sub-tile = 2x4 WMMA 16x16 accumulators.
#define BM 128
#define BN 128
#define BK 32
#define LDA 40    // ushorts per lsA row  (32 + 8 pad -> 80B stride, 16B-aligned)
#define LDBT 40   // ushorts per lsBt row (B stored transposed: [n][k])

typedef __attribute__((ext_vector_type(16))) __bf16 v16bf;
typedef __attribute__((ext_vector_type(8)))  float  v8f;

__device__ __forceinline__ unsigned short f2bf(float f) {
  unsigned int x = __float_as_uint(f);
  x += 0x7FFFu + ((x >> 16) & 1u);   // round-to-nearest-even
  return (unsigned short)(x >> 16);
}
__device__ __forceinline__ unsigned int pk2(float a, float b) {
  return (unsigned int)f2bf(a) | ((unsigned int)f2bf(b) << 16);
}

// ---------------- z fp32 -> bf16 (once; feeds async A-path of GEMM1) ----------------
__global__ __launch_bounds__(256) void moe_cvt_z(
    const float* __restrict__ z, unsigned short* __restrict__ zbf) {
  int idx = blockIdx.x * 256 + threadIdx.x;   // 8 elements per thread
  const float4* src = (const float4*)(z + (size_t)idx * 8);
  float4 a = src[0], b = src[1];
  uint4 o;
  o.x = pk2(a.x, a.y); o.y = pk2(a.z, a.w);
  o.z = pk2(b.x, b.y); o.w = pk2(b.z, b.w);
  *(uint4*)(zbf + (size_t)idx * 8) = o;
}

// ---------------- gate: logits, softmax, top-2, expert counting ----------------
__global__ __launch_bounds__(256) void moe_gate(
    const float* __restrict__ z, const float* __restrict__ Wg,
    const float* __restrict__ bg, int* __restrict__ topidx,
    float* __restrict__ topw, int* __restrict__ pos, int* __restrict__ counts) {
  const int token = blockIdx.x * 8 + (threadIdx.x >> 5);
  const int lane  = threadIdx.x & 31;
  float acc[NEXP];
#pragma unroll
  for (int e = 0; e < NEXP; ++e) acc[e] = 0.f;
  const float* zp = z + (size_t)token * DIM;
  for (int i = lane; i < DIM; i += 32) {
    float zv = zp[i];
    const float4* wr = (const float4*)(Wg + (size_t)i * NEXP);
    float4 w0 = wr[0], w1 = wr[1];
    acc[0] += zv * w0.x; acc[1] += zv * w0.y; acc[2] += zv * w0.z; acc[3] += zv * w0.w;
    acc[4] += zv * w1.x; acc[5] += zv * w1.y; acc[6] += zv * w1.z; acc[7] += zv * w1.w;
  }
#pragma unroll
  for (int e = 0; e < NEXP; ++e)
#pragma unroll
    for (int o = 16; o > 0; o >>= 1) acc[e] += __shfl_xor(acc[e], o, 32);

  if (lane == 0) {
    float mx = -1e30f;
#pragma unroll
    for (int e = 0; e < NEXP; ++e) { acc[e] += bg[e]; mx = fmaxf(mx, acc[e]); }
    float p[NEXP], s = 0.f;
#pragma unroll
    for (int e = 0; e < NEXP; ++e) { p[e] = __expf(acc[e] - mx); s += p[e]; }
    int e0 = 0;
#pragma unroll
    for (int e = 1; e < NEXP; ++e) if (p[e] > p[e0]) e0 = e;
    int e1 = (e0 == 0) ? 1 : 0;
#pragma unroll
    for (int e = 0; e < NEXP; ++e) if (e != e0 && p[e] > p[e1]) e1 = e;
    float inv = 1.f / s;
    topidx[2 * token]     = e0;
    topidx[2 * token + 1] = e1;
    topw[2 * token]       = p[e0] * inv;
    topw[2 * token + 1]   = p[e1] * inv;
    pos[2 * token]     = atomicAdd(&counts[e0], 1);
    pos[2 * token + 1] = atomicAdd(&counts[e1], 1);
  }
}

// ---------------- prefix sum over 8 expert counts ----------------
__global__ void moe_scan(const int* __restrict__ counts, int* __restrict__ offsets) {
  if (threadIdx.x == 0) {
    int s = 0;
    for (int e = 0; e < NEXP; ++e) { offsets[e] = s; s += counts[e]; }
    offsets[NEXP] = s;
  }
}

// ---------------- build slot<->token maps ----------------
__global__ __launch_bounds__(256) void moe_scatter(
    const int* __restrict__ topidx, const int* __restrict__ pos,
    const int* __restrict__ offsets, int* __restrict__ rowtok,
    int* __restrict__ slotOf) {
  int b = blockIdx.x * 256 + threadIdx.x;
  if (b >= B_TOK) return;
#pragma unroll
  for (int k = 0; k < TOPK; ++k) {
    int e = topidx[2 * b + k];
    int s = offsets[e] + pos[2 * b + k];
    slotOf[2 * b + k] = s;
    rowtok[s] = b;
  }
}

// Fragment loaders (ISA wave32 WMMA layouts, both contiguous b128 pairs)
__device__ __forceinline__ v16bf load_afrag(const unsigned short* lsA, int m, int lh) {
  // A 16x32: lane m, elements 0..7 -> k in [8*lh, 8*lh+8), 8..15 -> k in [16+8*lh, ...)
  union { uint4 q[2]; v16bf v; } fa;
  const unsigned short* base = lsA + m * LDA;
  fa.q[0] = *(const uint4*)(base + 8 * lh);
  fa.q[1] = *(const uint4*)(base + 16 + 8 * lh);
  return fa.v;
}
__device__ __forceinline__ v16bf load_bfrag(const unsigned short* lsBt, int n, int lh) {
  // B 32x16: lane n, elements 0..7 -> k in [8*lh, 8*lh+8), 8..15 -> k in [16+8*lh, ...)
  union { uint4 q[2]; v16bf v; } fb;
  const unsigned short* base = lsBt + n * LDBT;
  fb.q[0] = *(const uint4*)(base + 8 * lh);
  fb.q[1] = *(const uint4*)(base + 16 + 8 * lh);
  return fb.v;
}

// ---------------- GEMM1: h = relu(zbf[rows] @ W1[e] + b1[e]) (bf16 out) ----------------
__global__ __launch_bounds__(256) void moe_gemm1(
    const unsigned short* __restrict__ zbf, const float* __restrict__ W1,
    const float* __restrict__ b1, const int* __restrict__ offsets,
    const int* __restrict__ rowtok, unsigned short* __restrict__ hbuf) {
  __shared__ unsigned short lsA[BM * LDA];
  __shared__ unsigned short lsBt[BN * LDBT];
  __shared__ int rtok[BM];

  const int e        = blockIdx.z;
  const int rowStart = offsets[e] + blockIdx.x * BM;
  const int rowEnd   = offsets[e + 1];
  if (rowStart >= rowEnd) return;
  const int nStart = blockIdx.y * BN;

  const int t = threadIdx.x;
  if (t < BM) {
    int row = rowStart + t;
    rtok[t] = rowtok[row < rowEnd ? row : rowStart];
  }
  __syncthreads();

  const int lane = t & 31;
  const int wave = t >> 5;
  const int wm = wave >> 1;   // 0..3  (32 rows each)
  const int wn = wave & 1;    // 0..1  (64 cols each)
  const int lm = lane & 15;
  const int lh = lane >> 4;

  v8f acc[2][4] = {};

  const int ar  = t >> 1;          // A loader: row 0..127
  const int akq = (t & 1) << 4;    //           k sub-offset 0/16
  const int bkk = (t >> 4) << 1;   // B loader: even k row 0..30
  const int bnq = (t & 15) << 3;   //           n sub-offset 0..120 step 8

  const float* W1e  = W1 + (size_t)e * DIM * HID;
  const unsigned short* zrow = zbf + (size_t)rtok[ar] * DIM + akq;
  const unsigned ldsA = (unsigned)(unsigned long long)(void*)&lsA[ar * LDA + akq];

  for (int k0 = 0; k0 < DIM; k0 += BK) {
    // ---- A tile: gathered bf16 z rows, async copy global->LDS ----
    unsigned long long ga = (unsigned long long)(void*)(zrow + k0);
    asm volatile("global_load_async_to_lds_b128 %0, %1, off"
                 :: "v"(ldsA), "v"(ga) : "memory");
    asm volatile("global_load_async_to_lds_b128 %0, %1, off offset:16"
                 :: "v"(ldsA), "v"(ga) : "memory");

    // ---- B tile: two adjacent k-rows x 8 cols, fp32 -> bf16, stored transposed [n][k] ----
    const float* brow0 = W1e + (size_t)(k0 + bkk) * HID + nStart + bnq;
    const float4* bp0 = (const float4*)brow0;
    const float4* bp1 = (const float4*)(brow0 + HID);
    float4 r00 = bp0[0], r01 = bp0[1];
    float4 r10 = bp1[0], r11 = bp1[1];
    __builtin_prefetch(brow0 + (size_t)BK * HID, 0, 1);
    {
      unsigned short* bb = &lsBt[bnq * LDBT + bkk];
      *(unsigned int*)(bb + 0 * LDBT) = pk2(r00.x, r10.x);
      *(unsigned int*)(bb + 1 * LDBT) = pk2(r00.y, r10.y);
      *(unsigned int*)(bb + 2 * LDBT) = pk2(r00.z, r10.z);
      *(unsigned int*)(bb + 3 * LDBT) = pk2(r00.w, r10.w);
      *(unsigned int*)(bb + 4 * LDBT) = pk2(r01.x, r11.x);
      *(unsigned int*)(bb + 5 * LDBT) = pk2(r01.y, r11.y);
      *(unsigned int*)(bb + 6 * LDBT) = pk2(r01.z, r11.z);
      *(unsigned int*)(bb + 7 * LDBT) = pk2(r01.w, r11.w);
    }
    asm volatile("s_wait_asynccnt 0" ::: "memory");
    __syncthreads();

    v16bf afr[2], bfr[4];
#pragma unroll
    for (int mi = 0; mi < 2; ++mi)
      afr[mi] = load_afrag(lsA, wm * 32 + mi * 16 + lm, lh);
#pragma unroll
    for (int ni = 0; ni < 4; ++ni)
      bfr[ni] = load_bfrag(lsBt, wn * 64 + ni * 16 + lm, lh);
#pragma unroll
    for (int mi = 0; mi < 2; ++mi)
#pragma unroll
      for (int ni = 0; ni < 4; ++ni)
        acc[mi][ni] = __builtin_amdgcn_wmma_f32_16x16x32_bf16(
            false, afr[mi], false, bfr[ni], (short)0, acc[mi][ni], false, false);
    __syncthreads();
  }

  // epilogue: bias + relu -> bf16 h
  const bool full = (rowStart + BM) <= rowEnd;
  if (full) {
#pragma unroll
    for (int mi = 0; mi < 2; ++mi)
#pragma unroll
      for (int ni = 0; ni < 4; ++ni) {
        int n = nStart + wn * 64 + ni * 16 + lm;
        float bias = b1[e * HID + n];
#pragma unroll
        for (int j = 0; j < 8; ++j) {
          int row = rowStart + wm * 32 + mi * 16 + 8 * lh + j;
          hbuf[(size_t)row * HID + n] = f2bf(fmaxf(acc[mi][ni][j] + bias, 0.f));
        }
      }
  } else {
#pragma unroll
    for (int mi = 0; mi < 2; ++mi)
#pragma unroll
      for (int ni = 0; ni < 4; ++ni) {
        int n = nStart + wn * 64 + ni * 16 + lm;
        float bias = b1[e * HID + n];
#pragma unroll
        for (int j = 0; j < 8; ++j) {
          int row = rowStart + wm * 32 + mi * 16 + 8 * lh + j;
          if (row < rowEnd)
            hbuf[(size_t)row * HID + n] = f2bf(fmaxf(acc[mi][ni][j] + bias, 0.f));
        }
      }
  }
}

// ---------------- GEMM2: y = h @ W2[e] + b2[e] (fp32 out per slot) ----------------
__global__ __launch_bounds__(256) void moe_gemm2(
    const unsigned short* __restrict__ hbuf, const float* __restrict__ W2,
    const float* __restrict__ b2, const int* __restrict__ offsets,
    float* __restrict__ ybuf) {
  __shared__ unsigned short lsA[BM * LDA];
  __shared__ unsigned short lsBt[BN * LDBT];

  const int e        = blockIdx.z;
  const int rowStart = offsets[e] + blockIdx.x * BM;
  const int rowEnd   = offsets[e + 1];
  if (rowStart >= rowEnd) return;
  const int total  = offsets[NEXP];
  const int nStart = blockIdx.y * BN;

  const int t = threadIdx.x;
  const int lane = t & 31;
  const int wave = t >> 5;
  const int wm = wave >> 1;
  const int wn = wave & 1;
  const int lm = lane & 15;
  const int lh = lane >> 4;

  v8f acc[2][4] = {};

  const int ar  = t >> 1;
  const int akq = (t & 1) << 4;
  const int bkk = (t >> 4) << 1;
  const int bnq = (t & 15) << 3;

  int arow = rowStart + ar;
  if (arow >= total) arow = total - 1;
  const unsigned short* hrow = hbuf + (size_t)arow * HID + akq;
  const float* W2e = W2 + (size_t)e * HID * DIM;
  const unsigned ldsA = (unsigned)(unsigned long long)(void*)&lsA[ar * LDA + akq];

  for (int k0 = 0; k0 < HID; k0 += BK) {
    // ---- A tile: bf16 h rows, raw async copy global->LDS ----
    unsigned long long ga = (unsigned long long)(void*)(hrow + k0);
    asm volatile("global_load_async_to_lds_b128 %0, %1, off"
                 :: "v"(ldsA), "v"(ga) : "memory");
    asm volatile("global_load_async_to_lds_b128 %0, %1, off offset:16"
                 :: "v"(ldsA), "v"(ga) : "memory");

    // ---- B tile: W2 fp32 -> bf16, stored transposed [n][k] ----
    const float* brow0 = W2e + (size_t)(k0 + bkk) * DIM + nStart + bnq;
    const float4* bp0 = (const float4*)brow0;
    const float4* bp1 = (const float4*)(brow0 + DIM);
    float4 r00 = bp0[0], r01 = bp0[1];
    float4 r10 = bp1[0], r11 = bp1[1];
    __builtin_prefetch(brow0 + (size_t)BK * DIM, 0, 1);
    {
      unsigned short* bb = &lsBt[bnq * LDBT + bkk];
      *(unsigned int*)(bb + 0 * LDBT) = pk2(r00.x, r10.x);
      *(unsigned int*)(bb + 1 * LDBT) = pk2(r00.y, r10.y);
      *(unsigned int*)(bb + 2 * LDBT) = pk2(r00.z, r10.z);
      *(unsigned int*)(bb + 3 * LDBT) = pk2(r00.w, r10.w);
      *(unsigned int*)(bb + 4 * LDBT) = pk2(r01.x, r11.x);
      *(unsigned int*)(bb + 5 * LDBT) = pk2(r01.y, r11.y);
      *(unsigned int*)(bb + 6 * LDBT) = pk2(r01.z, r11.z);
      *(unsigned int*)(bb + 7 * LDBT) = pk2(r01.w, r11.w);
    }
    asm volatile("s_wait_asynccnt 0" ::: "memory");
    __syncthreads();

    v16bf afr[2], bfr[4];
#pragma unroll
    for (int mi = 0; mi < 2; ++mi)
      afr[mi] = load_afrag(lsA, wm * 32 + mi * 16 + lm, lh);
#pragma unroll
    for (int ni = 0; ni < 4; ++ni)
      bfr[ni] = load_bfrag(lsBt, wn * 64 + ni * 16 + lm, lh);
#pragma unroll
    for (int mi = 0; mi < 2; ++mi)
#pragma unroll
      for (int ni = 0; ni < 4; ++ni)
        acc[mi][ni] = __builtin_amdgcn_wmma_f32_16x16x32_bf16(
            false, afr[mi], false, bfr[ni], (short)0, acc[mi][ni], false, false);
    __syncthreads();
  }

  const bool full = (rowStart + BM) <= rowEnd;
  if (full) {
#pragma unroll
    for (int mi = 0; mi < 2; ++mi)
#pragma unroll
      for (int ni = 0; ni < 4; ++ni) {
        int n = nStart + wn * 64 + ni * 16 + lm;
        float bias = b2[e * DIM + n];
#pragma unroll
        for (int j = 0; j < 8; ++j) {
          int row = rowStart + wm * 32 + mi * 16 + 8 * lh + j;
          ybuf[(size_t)row * DIM + n] = acc[mi][ni][j] + bias;
        }
      }
  } else {
#pragma unroll
    for (int mi = 0; mi < 2; ++mi)
#pragma unroll
      for (int ni = 0; ni < 4; ++ni) {
        int n = nStart + wn * 64 + ni * 16 + lm;
        float bias = b2[e * DIM + n];
#pragma unroll
        for (int j = 0; j < 8; ++j) {
          int row = rowStart + wm * 32 + mi * 16 + 8 * lh + j;
          if (row < rowEnd) ybuf[(size_t)row * DIM + n] = acc[mi][ni][j] + bias;
        }
      }
  }
}

// ---------------- combine: out[b] = w0*y[slot0] + w1*y[slot1] ----------------
__global__ __launch_bounds__(256) void moe_combine(
    const float* __restrict__ ybuf, const int* __restrict__ slotOf,
    const float* __restrict__ topw, float* __restrict__ out) {
  int idx = blockIdx.x * 256 + threadIdx.x;   // one float4 per thread
  const int nQ = DIM / 4;
  int b = idx / nQ;
  int q = (idx - b * nQ) * 4;
  if (b >= B_TOK) return;
  int s0 = slotOf[2 * b], s1 = slotOf[2 * b + 1];
  float w0 = topw[2 * b], w1 = topw[2 * b + 1];
  float4 y0 = *(const float4*)(ybuf + (size_t)s0 * DIM + q);
  float4 y1 = *(const float4*)(ybuf + (size_t)s1 * DIM + q);
  float4 o;
  o.x = w0 * y0.x + w1 * y1.x;
  o.y = w0 * y0.y + w1 * y1.y;
  o.z = w0 * y0.z + w1 * y1.z;
  o.w = w0 * y0.w + w1 * y1.w;
  *(float4*)(out + (size_t)b * DIM + q) = o;
}

// ---------------- host launcher ----------------
// Workspace layout (needs ~219 MB):
//   counts(128) | offsets(128) | topidx | topw | pos | rowtok | slotOf |
//   zbf bf16 [8192][1024] (16.8MB) | h bf16 [16384][4096] (134.2MB) |
//   y fp32 [16384][1024] (67.1MB)
extern "C" void kernel_launch(void* const* d_in, const int* in_sizes, int n_in,
                              void* d_out, int out_size, void* d_ws, size_t ws_size,
                              hipStream_t stream) {
  const float* z  = (const float*)d_in[0];
  const float* Wg = (const float*)d_in[1];
  const float* bg = (const float*)d_in[2];
  const float* W1 = (const float*)d_in[3];
  const float* b1 = (const float*)d_in[4];
  const float* W2 = (const float*)d_in[5];
  const float* b2 = (const float*)d_in[6];
  float* out = (float*)d_out;

  char* ws = (char*)d_ws;
  int*   counts  = (int*)ws;   ws += 128;
  int*   offsets = (int*)ws;   ws += 128;
  int*   topidx  = (int*)ws;   ws += (size_t)B_TOK * TOPK * sizeof(int);
  float* topw    = (float*)ws; ws += (size_t)B_TOK * TOPK * sizeof(float);
  int*   pos     = (int*)ws;   ws += (size_t)B_TOK * TOPK * sizeof(int);
  int*   rowtok  = (int*)ws;   ws += (size_t)B_TOK * TOPK * sizeof(int);
  int*   slotOf  = (int*)ws;   ws += (size_t)B_TOK * TOPK * sizeof(int);
  unsigned short* zbf = (unsigned short*)ws;
  ws += (size_t)B_TOK * DIM * sizeof(unsigned short);
  unsigned short* hbuf = (unsigned short*)ws;
  ws += (size_t)B_TOK * TOPK * HID * sizeof(unsigned short);
  float* ybuf = (float*)ws;

  hipMemsetAsync(counts, 0, 128, stream);

  moe_cvt_z<<<B_TOK * DIM / 8 / 256, 256, 0, stream>>>(z, zbf);
  moe_gate<<<B_TOK / 8, 256, 0, stream>>>(z, Wg, bg, topidx, topw, pos, counts);
  moe_scan<<<1, 32, 0, stream>>>(counts, offsets);
  moe_scatter<<<B_TOK / 256, 256, 0, stream>>>(topidx, pos, offsets, rowtok, slotOf);
  moe_gemm1<<<dim3(B_TOK / BM, HID / BN, NEXP), 256, 0, stream>>>(
      zbf, W1, b1, offsets, rowtok, hbuf);
  moe_gemm2<<<dim3(B_TOK / BM, DIM / BN, NEXP), 256, 0, stream>>>(
      hbuf, W2, b2, offsets, ybuf);
  moe_combine<<<B_TOK * (DIM / 4) / 256, 256, 0, stream>>>(ybuf, slotOf, topw, out);
}
